// kernel_generated_1_62904091017265
// MI455X (gfx1250) — compile-verified, matrix operand order checked by reading, not run on hardware
//
#include <hip/hip_runtime.h>

// ---------------------------------------------------------------------------
// Shifted-group-mix 3x3 conv as implicit GEMM on CDNA5 WMMA (bf16 -> f32).
//   out[n,i,h,w] = sum_{l,j,k} W[l,j,k,i] * A[(n,h,w),(l,j,k)]
// K permuted to (j,k,l): one K-step per (j,k), 32 l-channels per WMMA.
// x halo staged into LDS with GLOBAL_LOAD_ASYNC_TO_LDS_B128 (ASYNCcnt),
// A fragments load from LDS tensor s2[hh][w][k][l] (conflict-free b128),
// B fragments load from a globally pre-packed bf16 image of W (L2-resident)
// -> the 63-WMMA main loop is barrier-free.
// ---------------------------------------------------------------------------

typedef __attribute__((ext_vector_type(16))) __bf16 v16bf;
typedef __attribute__((ext_vector_type(8)))  __bf16 v8bf;
typedef __attribute__((ext_vector_type(8)))  float  v8f;

#define NIMG   1024
#define HDIM   28
#define WDIM   28
#define COUT   128
#define BH     4                     // h rows per band
#define NBAND  7                     // 28 / BH
#define SROWS  6                     // BH + 2 halo rows

#define S2_STRIDE  104               // halfs per (hh,w) cell: 3*32 data + 8 pad
                                     // 208 B: 16B-aligned frags, 52-dword lane
                                     // stride -> conflict-free ds_load_b128
#define SZ_S2      (SROWS * WDIM * S2_STRIDE * 2)   // 34944
#define OFF_HALO   SZ_S2                             // 34944 (16B aligned)
#define SZ_HALO    (64 * 3 * WDIM * 4)               // 21504 (3-row stage)
#define CT_STRIDE  116                               // floats, pad for banks
#define SMEM_BYTES (COUT * CT_STRIDE * 4)            // 59392 >= 56448 main

#define BP_ELEMS   (9 * COUT * 32)                   // 36864 bf16 = 73728 B

// ---- Pre-pack: Bp[(s*128 + i)*32 + l] = bf16( W[(l*9 + s)*128 + i] ) ------
__global__ __launch_bounds__(256)
void pack_w_bf16(const float* __restrict__ Wg, __bf16* __restrict__ Bp)
{
    int e = blockIdx.x * 256 + threadIdx.x;   // 36864 elements
    int l = e & 31;
    int t = e >> 5;
    int i = t & 127;
    int s = t >> 7;                           // 0..8  (= j*3 + k)
    Bp[e] = (__bf16)Wg[(l * 9 + s) * COUT + i];
}

__global__ __launch_bounds__(256)
void shiftconv_wmma_bf16(const float* __restrict__ x,
                         const __bf16* __restrict__ Bp,
                         float* __restrict__ out)
{
    __shared__ __attribute__((aligned(16))) unsigned char smem[SMEM_BYTES];
    __bf16* s2 = (__bf16*)(smem);               // s2[hh][w][k][l]
    float*  hb = (float*)(smem + OFF_HALO);     // halo[c][3][28] f32
    float*  sC = (float*)(smem);                // epilogue tile (aliases)

    const int tid  = threadIdx.x;
    const int lane = tid & 31;
    const int wv   = tid >> 5;               // 8 waves, one 16-col tile each
    const int n    = blockIdx.x / NBAND;     // image-major: x hot in L2
    const int band = blockIdx.x % NBAND;
    const int h0   = band * BH;

    const int khalf = lane >> 4;             // WMMA K-half selector
    const int r16   = lane & 15;             // row (A) / col (B) in 16-tile

    if (tid == 0)                            // keep Bp warm (global_prefetch)
        __builtin_prefetch(Bp, 0, 0);

    const float* xn = x + (size_t)n * 64 * HDIM * WDIM;

    // ---- Phase 0: two 3-row stages: async halo copy -> combine into s2 ----
    #pragma unroll
    for (int st = 0; st < 2; ++st) {
        const int hh0 = st * 3;
        // (a) async bulk copy: 1344 aligned 16B chunks, lds addr = q*16.
        //     3 consecutive h rows per channel are contiguous (84 floats).
        if (tid < 224) {
            #pragma unroll
            for (int it = 0; it < 6; ++it) {
                int q   = tid + it * 224;        // 0..1343
                int c   = q / 21;                // channel
                int o   = q - c * 21;            // float4 within 84-float blk
                int rl  = o / 7;                 // local row 0..2
                int wc4 = o - rl * 7;            // float4 within row
                int hp  = h0 - 1 + hh0 + rl;
                int hps = hp < 0 ? 0 : (hp > 27 ? 27 : hp);  // safe row;
                const float* ga = xn + c * 784 + hps * 28 + wc4 * 4;
                unsigned lo = (unsigned)(OFF_HALO + q * 16);
                asm volatile("global_load_async_to_lds_b128 %0, %1, off"
                             :: "v"(lo), "v"(ga) : "memory");
            }
        }
        asm volatile("s_wait_asynccnt 0x0" ::: "memory");
        __syncthreads();

        // (b) combine pair of channels (l, 32+l) with width masks -> s2.
        //     168 threads: 84 (hh_loc,w) cells x 2 l-halves; l in pairs so
        //     converts pack (v_cvt_pk_bf16_f32) and stores are b32.
        if (tid < 168) {
            const int lh     = tid / 84;           // l-half: 0 or 1
            const int cell   = tid - lh * 84;
            const int hh_loc = cell / WDIM;
            const int w      = cell % WDIM;
            const int hh     = hh0 + hh_loc;
            const int hp     = h0 - 1 + hh;
            const bool hv    = (hp >= 0) && (hp < HDIM);
            const int w27 = (w + 27) % 28, w26 = (w + 26) % 28,
                      w25 = (w + 25) % 28;
            const bool m10 = (w != 0), m12 = (w != 27);  // term1 k=0,2
            const bool m20 = (w != 1), m22 = (w != 0);   // term2 k=0,2
            __bf16* sc = s2 + (hh * WDIM + w) * S2_STRIDE;
            const float* f1 = hb + (lh * 16) * 84 + hh_loc * 28;       // ch l
            const float* f2 = hb + (32 + lh * 16) * 84 + hh_loc * 28;  // 32+l
            #pragma unroll 2
            for (int l2 = 0; l2 < 16; l2 += 2) {
                int l = lh * 16 + l2;
                float v0a = 0.f, v1a = 0.f, v2a = 0.f;
                float v0b = 0.f, v1b = 0.f, v2b = 0.f;
                if (hv) {
                    float a0 = f1[w26], a1 = f1[w27], a2 = f1[w];
                    float b0 = f2[w25], b1 = f2[w26], b2 = f2[w27];
                    v0a = (m10 ? a0 : 0.f) + (m20 ? b0 : 0.f);
                    v1a = a1 + b1;                       // k=1 always valid
                    v2a = (m12 ? a2 : 0.f) + (m22 ? b2 : 0.f);
                    float c0 = f1[84 + w26], c1 = f1[84 + w27], c2 = f1[84 + w];
                    float d0 = f2[84 + w25], d1 = f2[84 + w26], d2 = f2[84 + w27];
                    v0b = (m10 ? c0 : 0.f) + (m20 ? d0 : 0.f);
                    v1b = c1 + d1;
                    v2b = (m12 ? c2 : 0.f) + (m22 ? d2 : 0.f);
                }
                union { __bf16 h[2]; unsigned u; } q0, q1, q2;
                q0.h[0] = (__bf16)v0a; q0.h[1] = (__bf16)v0b;
                q1.h[0] = (__bf16)v1a; q1.h[1] = (__bf16)v1b;
                q2.h[0] = (__bf16)v2a; q2.h[1] = (__bf16)v2b;
                *(unsigned*)(sc + l)      = q0.u;
                *(unsigned*)(sc + 32 + l) = q1.u;
                *(unsigned*)(sc + 64 + l) = q2.u;
                f1 += 2 * 84; f2 += 2 * 84;
            }
        }
        __syncthreads();                 // stage done (halo reusable / s2 ok)
    }

    v8f acc[7];
    #pragma unroll
    for (int mt = 0; mt < 7; ++mt)
        #pragma unroll
        for (int g = 0; g < 8; ++g) acc[mt][g] = 0.0f;

    // ---- Main loop: 9 K-steps (j,k), fully unrolled, barrier-free ---------
    #pragma unroll
    for (int s = 0; s < 9; ++s) {
        const int j = s / 3, k = s % 3;      // compile-time after unroll
        union Frag { v16bf v; struct { v8bf lo, hi; } h; } bf, af;
        {   // B fragment: two global_load_b128 from L2-resident pack
            const __bf16* pb =
                Bp + ((size_t)(s * COUT + 16 * wv + r16) << 5) + khalf * 8;
            bf.h.lo = *(const v8bf*)pb;
            bf.h.hi = *(const v8bf*)(pb + 16);
        }
        #pragma unroll
        for (int mt = 0; mt < 7; ++mt) {
            // m = 16*mt + r16 = dh*28+w  ->  cell (dh+j, w) = m + j*28
            const __bf16* pa =
                s2 + (16 * mt + r16 + j * WDIM) * S2_STRIDE + k * 32 + khalf * 8;
            af.h.lo = *(const v8bf*)pa;
            af.h.hi = *(const v8bf*)(pa + 16);
            acc[mt] = __builtin_amdgcn_wmma_f32_16x16x32_bf16(
                false, af.v, false, bf.v, (short)0, acc[mt], false, false);
        }
    }

    // ---- Epilogue: acc -> LDS (conflict-free) -> coalesced b128 stores ----
    __syncthreads();                         // all A-frag reads done (alias)
    #pragma unroll
    for (int mt = 0; mt < 7; ++mt)
        #pragma unroll
        for (int g = 0; g < 8; ++g) {
            int row = 16 * mt + g + 8 * khalf;       // m within band tile
            int col = 16 * wv + r16;                 // output channel i
            sC[col * CT_STRIDE + row] = acc[mt][g];
        }
    __syncthreads();

    // (h0+dh)*28 + w0 == h0*28 + m0  -> store address linear in m0, no divs.
    {
        const int i2    = tid >> 1;                  // output channel
        const int mbase = (tid & 1) * 56;            // half of the 112 m's
        const float* srcp = sC + i2 * CT_STRIDE + mbase;
        float* dstp = out + (size_t)n * COUT * HDIM * WDIM
                          + (size_t)i2 * HDIM * WDIM + h0 * WDIM + mbase;
        #pragma unroll
        for (int it = 0; it < 14; ++it)
            *(float4*)(dstp + it * 4) = *(const float4*)(srcp + it * 4);
    }
}

extern "C" void kernel_launch(void* const* d_in, const int* in_sizes, int n_in,
                              void* d_out, int out_size, void* d_ws, size_t ws_size,
                              hipStream_t stream) {
    (void)in_sizes; (void)n_in; (void)out_size; (void)ws_size;
    const float* x = (const float*)d_in[0];   // (1024, 64, 28, 28) fp32
    const float* W = (const float*)d_in[1];   // (32, 3, 3, 128)    fp32
    float* out = (float*)d_out;               // (1024, 128, 28, 28) fp32
    __bf16* Bp = (__bf16*)d_ws;               // 73728 B fragment-ordered pack

    pack_w_bf16<<<BP_ELEMS / 256, 256, 0, stream>>>(W, Bp);
    shiftconv_wmma_bf16<<<NIMG * NBAND, 256, 0, stream>>>(x, Bp, out);
}
